// GraphAttentionLayer_70308614635892
// MI455X (gfx1250) — matrix operand order repaired and checked by reading, not compiled, not run on hardware
//
#include <hip/hip_runtime.h>
#include <hip/hip_bf16.h>
#include <stdint.h>

// GAT layer, fused for MI455X (gfx1250, wave32, WMMA).
//   B=64, N=1024, F_IN=256, F_OUT=64
// Pipeline:
//   k_sum_a1 : S = sum(a[0:64])
//   k_feat   : h = input@W via split-bf16 WMMA (3-mult hi/lo, ~fp32 accurate);
//              writes hT_hi/hT_lo (bf16, transposed [b][n][j]) + exact e2[b][j].
//   k_attn   : flash-style fused softmax(adj-masked relu(S*h[i,j>>4]+e2[j])) @ h
//              with online sum (no max needed: relu => e in [0,~40], exp safe in f32),
//              PV via 12x v_wmma_f32_16x16x32_bf16 per 32-wide j block per wave,
//              per-batch hT planes staged in 260KB LDS, elu epilogue.
// Cache policy: adj/input loads and out stores are non-temporal (streamed once,
// 256MB adj > 192MB L2); hT planes (16MB) stay regular-temporal so they remain
// L2-resident across k_attn workgroups.

#define B_DIM 64
#define N_DIM 1024
#define F_IN  256
#define F_OUT 64

typedef __attribute__((ext_vector_type(16))) __bf16 v16bf;
typedef __attribute__((ext_vector_type(8)))  float  v8f;
typedef __attribute__((ext_vector_type(4)))  float  f32x4;  // native ext vector for NT loads
typedef __attribute__((ext_vector_type(4)))  int    i32x4;  // native ext vector for NT loads

struct U32B { uint4 a; uint4 b; };

static __device__ __forceinline__ v16bf mk_v16bf(uint4 a, uint4 b) {
  U32B u; u.a = a; u.b = b;
  return __builtin_bit_cast(v16bf, u);
}
static __device__ __forceinline__ unsigned short f2bf_bits(float f) {
  __bf16 h = (__bf16)f;
  return __builtin_bit_cast(unsigned short, h);
}
static __device__ __forceinline__ float bfbits2f(unsigned short u) {
  __bf16 h = __builtin_bit_cast(__bf16, u);
  return (float)h;
}

// ---------------------------------------------------------------- kernel 0
__global__ void k_sum_a1(const float* __restrict__ a, float* __restrict__ Sout) {
  __shared__ float sa[64];
  int t = threadIdx.x;
  sa[t] = a[t];
  __syncthreads();
  if (t == 0) {
    float s = 0.f;
    for (int i = 0; i < 64; ++i) s += sa[i];
    Sout[0] = s;
  }
}

// ---------------------------------------------------------------- kernel 1
// h = input @ W ; 16 rows per wave, 4 waves per block.
__global__ void __launch_bounds__(128)
k_feat(const float* __restrict__ input, const float* __restrict__ W,
       const float* __restrict__ a,
       unsigned short* __restrict__ hT_hi, unsigned short* __restrict__ hT_lo,
       float* __restrict__ e2out) {
  __shared__ unsigned short WThi[F_OUT * F_IN];   // 32 KB, W^T hi plane
  __shared__ unsigned short WTlo[F_OUT * F_IN];   // 32 KB, W^T lo plane

  const int tid = threadIdx.x;
  for (int idx = tid; idx < F_IN * F_OUT; idx += 128) {
    const int k = idx >> 6, n = idx & 63;         // W[k][n], coalesced read
    const float w = W[idx];
    const unsigned short hb = f2bf_bits(w);
    WThi[n * F_IN + k] = hb;
    WTlo[n * F_IN + k] = f2bf_bits(w - bfbits2f(hb));
  }
  __syncthreads();

  const int wv   = tid >> 5;
  const int lane = tid & 31;
  const int m    = lane & 15;
  const int hi   = lane >> 4;
  const int kb   = hi * 8;

  const int row0 = (blockIdx.x * 4 + wv) * 16;    // global row tile over B*N
  const int b    = row0 >> 10;
  const int ii   = row0 & 1023;

  const float* inp = input + (size_t)(row0 + m) * F_IN;

  v8f acc[4];
  #pragma unroll
  for (int nt = 0; nt < 4; ++nt)
    #pragma unroll
    for (int v = 0; v < 8; ++v) acc[nt][v] = 0.f;

  for (int kblk = 0; kblk < F_IN / 32; ++kblk) {
    const int k0 = kblk * 32;
    // A operand (16x32 bf16): lane m holds k in {kb..kb+7} U {16+kb..16+kb+7}
    f32x4 f0 = __builtin_nontemporal_load((const f32x4*)(inp + k0 + kb));
    f32x4 f1 = __builtin_nontemporal_load((const f32x4*)(inp + k0 + kb + 4));
    f32x4 f2 = __builtin_nontemporal_load((const f32x4*)(inp + k0 + 16 + kb));
    f32x4 f3 = __builtin_nontemporal_load((const f32x4*)(inp + k0 + 16 + kb + 4));
    const float fs[16] = {f0.x,f0.y,f0.z,f0.w, f1.x,f1.y,f1.z,f1.w,
                          f2.x,f2.y,f2.z,f2.w, f3.x,f3.y,f3.z,f3.w};
    v16bf Ahi, Alo;
    #pragma unroll
    for (int t = 0; t < 16; ++t) {
      __bf16 h = (__bf16)fs[t];
      Ahi[t] = h;
      Alo[t] = (__bf16)(fs[t] - (float)h);
    }
    #pragma unroll
    for (int nt = 0; nt < 4; ++nt) {
      // B operand (32x16 bf16): lane n holds k in {kb..kb+7} U {16+kb..16+kb+7}
      const int baddr = (nt * 16 + m) * F_IN + k0 + hi * 8;
      uint4 q0 = *(const uint4*)(WThi + baddr);
      uint4 q1 = *(const uint4*)(WThi + baddr + 16);
      uint4 r0 = *(const uint4*)(WTlo + baddr);
      uint4 r1 = *(const uint4*)(WTlo + baddr + 16);
      v16bf Bh = mk_v16bf(q0, q1);
      v16bf Bl = mk_v16bf(r0, r1);
      acc[nt] = __builtin_amdgcn_wmma_f32_16x16x32_bf16(false, Ahi, false, Bh, (short)0, acc[nt], false, false);
      acc[nt] = __builtin_amdgcn_wmma_f32_16x16x32_bf16(false, Ahi, false, Bl, (short)0, acc[nt], false, false);
      acc[nt] = __builtin_amdgcn_wmma_f32_16x16x32_bf16(false, Alo, false, Bh, (short)0, acc[nt], false, false);
    }
  }

  // e2[row] = h[row,:] . a2  (exact f32 from accumulators)
  float part[8];
  #pragma unroll
  for (int v = 0; v < 8; ++v) part[v] = 0.f;
  #pragma unroll
  for (int nt = 0; nt < 4; ++nt) {
    const float a2v = a[F_OUT + nt * 16 + m];
    #pragma unroll
    for (int v = 0; v < 8; ++v) part[v] += acc[nt][v] * a2v;
  }
  #pragma unroll
  for (int v = 0; v < 8; ++v) {
    #pragma unroll
    for (int off = 1; off < 16; off <<= 1)
      part[v] += __shfl_xor(part[v], off, 32);   // reduce over n within half
  }
  if (m == 0) {
    #pragma unroll
    for (int v = 0; v < 8; ++v)
      e2out[row0 + hi * 8 + v] = part[v];        // C row = v + 8*hi
  }

  // Transposed packed bf16 hi/lo stores: hT[b][n][i], 8 consecutive i per lane.
  // Regular-temporal: 16MB total, re-read 4x by k_attn -> keep L2-resident.
  #pragma unroll
  for (int nt = 0; nt < 4; ++nt) {
    const int n = nt * 16 + m;
    unsigned int hw[8], lw[8];
    #pragma unroll
    for (int v = 0; v < 8; ++v) {
      const float x = acc[nt][v];
      const unsigned short hb = f2bf_bits(x);
      hw[v] = hb;
      lw[v] = f2bf_bits(x - bfbits2f(hb));
    }
    uint4 ph, pl;
    ph.x = hw[0] | (hw[1] << 16); ph.y = hw[2] | (hw[3] << 16);
    ph.z = hw[4] | (hw[5] << 16); ph.w = hw[6] | (hw[7] << 16);
    pl.x = lw[0] | (lw[1] << 16); pl.y = lw[2] | (lw[3] << 16);
    pl.z = lw[4] | (lw[5] << 16); pl.w = lw[6] | (lw[7] << 16);
    const size_t off = ((size_t)(b * F_OUT + n) * N_DIM) + ii + hi * 8;
    *(uint4*)(hT_hi + off) = ph;
    *(uint4*)(hT_lo + off) = pl;
  }
}

// ---------------------------------------------------------------- kernel 2
// Fused masked-softmax + attention@h + elu. 16 waves/block, 256 rows/block.
__global__ void __launch_bounds__(512)
k_attn(const int* __restrict__ adj, const float* __restrict__ e2g,
       const float* __restrict__ Sp,
       const unsigned short* __restrict__ hT_hi,
       const unsigned short* __restrict__ hT_lo,
       float* __restrict__ out) {
  extern __shared__ char smem[];
  unsigned short* lds_hi = (unsigned short*)smem;        // 128 KB
  unsigned short* lds_lo = lds_hi + F_OUT * N_DIM;       // 128 KB
  float*          lds_e2 = (float*)(lds_lo + F_OUT * N_DIM); // 4 KB

  const int tid = threadIdx.x;
  const int b   = blockIdx.x >> 2;
  const int iwg = (blockIdx.x & 3) * 256;

  { // stage this batch's hT planes + e2 into LDS (should hit L2)
    const uint4* sh = (const uint4*)(hT_hi + (size_t)b * F_OUT * N_DIM);
    const uint4* sl = (const uint4*)(hT_lo + (size_t)b * F_OUT * N_DIM);
    uint4* dh = (uint4*)lds_hi;
    uint4* dl = (uint4*)lds_lo;
    for (int idx = tid; idx < F_OUT * N_DIM / 8; idx += 512) {
      dh[idx] = sh[idx];
      dl[idx] = sl[idx];
    }
    for (int idx = tid; idx < N_DIM; idx += 512)
      lds_e2[idx] = e2g[b * N_DIM + idx];
  }
  const float S = Sp[0];
  __syncthreads();

  const int wv   = tid >> 5;
  const int lane = tid & 31;
  const int m    = lane & 15;
  const int hi   = lane >> 4;
  const int kb   = hi * 8;
  const int i0   = iwg + wv * 16;
  const int irow = i0 + m;

  const int* adjrow = adj + (((size_t)b * N_DIM) + irow) * N_DIM;

  v8f acc[4];
  #pragma unroll
  for (int nt = 0; nt < 4; ++nt)
    #pragma unroll
    for (int v = 0; v < 8; ++v) acc[nt][v] = 0.f;
  float psum = 0.f;

  for (int j0 = 0; j0 < N_DIM; j0 += 32) {
    // adj: 256MB streamed exactly once -> non-temporal, keep L2 for hT
    i32x4 A0 = __builtin_nontemporal_load((const i32x4*)(adjrow + j0 + kb));
    i32x4 A1 = __builtin_nontemporal_load((const i32x4*)(adjrow + j0 + kb + 4));
    i32x4 A2 = __builtin_nontemporal_load((const i32x4*)(adjrow + j0 + 16 + kb));
    i32x4 A3 = __builtin_nontemporal_load((const i32x4*)(adjrow + j0 + 16 + kb + 4));
    float4 E0 = *(const float4*)(lds_e2 + j0 + kb);
    float4 E1 = *(const float4*)(lds_e2 + j0 + kb + 4);
    float4 E2 = *(const float4*)(lds_e2 + j0 + 16 + kb);
    float4 E3 = *(const float4*)(lds_e2 + j0 + 16 + kb + 4);

    const int c0 = j0 >> 4;   // j>>4 is constant within each 16-wide half
    const float ga = S * (bfbits2f(lds_hi[c0 * N_DIM + irow]) +
                          bfbits2f(lds_lo[c0 * N_DIM + irow]));
    const float gb = S * (bfbits2f(lds_hi[(c0 + 1) * N_DIM + irow]) +
                          bfbits2f(lds_lo[(c0 + 1) * N_DIM + irow]));

    float p[16];
#define GAT_P(idx, Ai, Ei, g) { float e_ = (g) + (Ei); e_ = fmaxf(e_, 0.f); \
                                p[idx] = ((Ai) > 0) ? __expf(e_) : 0.f; }
    GAT_P(0,  A0.x, E0.x, ga) GAT_P(1,  A0.y, E0.y, ga)
    GAT_P(2,  A0.z, E0.z, ga) GAT_P(3,  A0.w, E0.w, ga)
    GAT_P(4,  A1.x, E1.x, ga) GAT_P(5,  A1.y, E1.y, ga)
    GAT_P(6,  A1.z, E1.z, ga) GAT_P(7,  A1.w, E1.w, ga)
    GAT_P(8,  A2.x, E2.x, gb) GAT_P(9,  A2.y, E2.y, gb)
    GAT_P(10, A2.z, E2.z, gb) GAT_P(11, A2.w, E2.w, gb)
    GAT_P(12, A3.x, E3.x, gb) GAT_P(13, A3.y, E3.y, gb)
    GAT_P(14, A3.z, E3.z, gb) GAT_P(15, A3.w, E3.w, gb)
#undef GAT_P

    v16bf Phi, Plo;
    float ps = 0.f;
    #pragma unroll
    for (int t = 0; t < 16; ++t) {
      ps += p[t];
      __bf16 h = (__bf16)p[t];
      Phi[t] = h;
      Plo[t] = (__bf16)(p[t] - (float)h);
    }
    psum += ps;

    #pragma unroll
    for (int nt = 0; nt < 4; ++nt) {
      const int baddr = (nt * 16 + m) * N_DIM + j0 + hi * 8;
      uint4 q0 = *(const uint4*)(lds_hi + baddr);
      uint4 q1 = *(const uint4*)(lds_hi + baddr + 16);
      uint4 r0 = *(const uint4*)(lds_lo + baddr);
      uint4 r1 = *(const uint4*)(lds_lo + baddr + 16);
      v16bf Bh = mk_v16bf(q0, q1);
      v16bf Bl = mk_v16bf(r0, r1);
      acc[nt] = __builtin_amdgcn_wmma_f32_16x16x32_bf16(false, Phi, false, Bh, (short)0, acc[nt], false, false);
      acc[nt] = __builtin_amdgcn_wmma_f32_16x16x32_bf16(false, Phi, false, Bl, (short)0, acc[nt], false, false);
      acc[nt] = __builtin_amdgcn_wmma_f32_16x16x32_bf16(false, Plo, false, Bh, (short)0, acc[nt], false, false);
    }
  }

  // row softmax denominators: lanes m and m+16 jointly cover row i0+m
  const float rowsum = psum + __shfl_xor(psum, 16, 32);

  #pragma unroll
  for (int v = 0; v < 8; ++v) {
    const float rs   = __shfl(rowsum, v + 8 * hi, 32);  // C row = v + 8*hi
    const float invr = 1.0f / fmaxf(rs, 1e-30f);
    float* orow = out + ((size_t)b * N_DIM + i0 + 8 * hi + v) * F_OUT;
    #pragma unroll
    for (int nt = 0; nt < 4; ++nt) {
      float x = acc[nt][v] * invr;
      x = (x > 0.f) ? x : (__expf(x) - 1.0f);           // elu(alpha=1)
      __builtin_nontemporal_store(x, orow + nt * 16 + m); // written once
    }
  }
}

// ---------------------------------------------------------------- launcher
extern "C" void kernel_launch(void* const* d_in, const int* in_sizes, int n_in,
                              void* d_out, int out_size, void* d_ws, size_t ws_size,
                              hipStream_t stream) {
  (void)in_sizes; (void)n_in; (void)out_size; (void)ws_size;
  const float* input = (const float*)d_in[0];
  const int*   adj   = (const int*)d_in[1];
  const float* W     = (const float*)d_in[2];
  const float* a     = (const float*)d_in[3];
  float* out = (float*)d_out;

  char* ws = (char*)d_ws;
  unsigned short* hT_hi = (unsigned short*)ws;                           // 8 MB
  unsigned short* hT_lo = (unsigned short*)(ws + (size_t)8 * 1024 * 1024);   // 8 MB
  float* e2 = (float*)(ws + (size_t)16 * 1024 * 1024);                   // 256 KB
  float* Sp = (float*)(ws + (size_t)16 * 1024 * 1024 + 256 * 1024);      // 4 B

  k_sum_a1<<<1, 64, 0, stream>>>(a, Sp);

  // B*N rows, 16 per wave, 4 waves per block -> 1024 blocks
  k_feat<<<(B_DIM * N_DIM) / (16 * 4), 128, 0, stream>>>(input, W, a, hT_hi, hT_lo, e2);

  // 64 batches x 4 i-chunks of 256 rows; 260 KB dynamic LDS per workgroup
  const size_t lds2 = (size_t)2 * F_OUT * N_DIM * 2 + N_DIM * 4; // 266240 B
  k_attn<<<B_DIM * (N_DIM / 256), 512, lds2, stream>>>(adj, e2, Sp, hT_hi, hT_lo, out);
}
